// APNet2_MPNN_78400333021551
// MI455X (gfx1250) — compile-verified, matrix operand order checked by reading, not run on hardware
//
#include <hip/hip_runtime.h>

typedef __attribute__((ext_vector_type(16))) _Float16 v16h;
typedef __attribute__((ext_vector_type(8)))  float    v8f;

// ---------------------------------------------------------------------------
// Packed-weight sizes (halfs).  Hidden ("hid") MLP: 296->256->128->64->8
// (K padded 296->320, final N padded 8->16).  Readout ("ro") MLP:
// 122->256->128->64->1 (K padded 122->128, final N padded 1->16).
// ---------------------------------------------------------------------------
constexpr int HID_W0 = 320 * 256;
constexpr int HID_W1 = 256 * 128;
constexpr int HID_W2 = 128 * 64;
constexpr int HID_W3 = 64 * 16;
constexpr int HID_TOT = HID_W0 + HID_W1 + HID_W2 + HID_W3;
constexpr int RO_W0 = 128 * 256;
constexpr int RO_W1 = 256 * 128;
constexpr int RO_W2 = 128 * 64;
constexpr int RO_W3 = 64 * 16;
constexpr int RO_TOT = RO_W0 + RO_W1 + RO_W2 + RO_W3;

struct RoArgs {
  const float* b0[4];
  const float* b1[4];
  const float* b2[4];
  const float* b3[4];
};

// ---------------------------------------------------------------------------
// WMMA fragment helpers (wave32, v_wmma_f32_16x16x32_f16)
// A (16x32 f16): lanes 0-15 hold M=lane with K 0-7 then 16-23;
//                lanes 16-31 hold M=lane-16 with K 8-15 then 24-31.
// B packed by the prep kernel so each lane reads 16 contiguous halfs.
// D (16x16 f32): lane holds col = lane&15; VGPR v holds row v (+8 if lane>=16).
// ---------------------------------------------------------------------------
__device__ inline v16h load_a(const _Float16* __restrict__ p, int ld, int lane) {
  const int row = lane & 15;
  const int kb = (lane >> 4) << 3;
  const _Float16* r = p + row * ld;
  v16h a;
#pragma unroll
  for (int j = 0; j < 8; ++j) a[j] = r[kb + j];
#pragma unroll
  for (int j = 0; j < 8; ++j) a[8 + j] = r[16 + kb + j];
  return a;
}

__device__ inline v16h load_b_packed(const _Float16* __restrict__ p, int lane) {
  const _Float16* q = p + (lane << 4);
  v16h b;
#pragma unroll
  for (int j = 0; j < 16; ++j) b[j] = q[j];
  return b;
}

__device__ inline v8f wmma_acc(v16h a, v16h b, v8f c) {
  return __builtin_amdgcn_wmma_f32_16x16x32_f16(false, a, false, b, (short)0, c,
                                                false, false);
}

// One dense layer: LDS f16 in [MT*16][KPAD] -> bias+ReLU -> LDS f16 out
// [MT*16][NPAD].  Weights packed fragment-major: tile (kt,nt) occupies 512
// contiguous halfs.
template <int KPAD, int NPAD, int MT>
__device__ inline void mlp_layer(const _Float16* __restrict__ inLds,
                                 const _Float16* __restrict__ Wp,
                                 const float* __restrict__ bias, int realN,
                                 _Float16* __restrict__ outLds, int lane) {
  const int col = lane & 15;
  const int rb = (lane >> 4) << 3;
  constexpr int NTN = NPAD >> 4;
  constexpr int KTN = KPAD >> 5;
  for (int mt = 0; mt < MT; ++mt) {
    const _Float16* inRow = inLds + mt * 16 * KPAD;
    for (int nt = 0; nt < NTN; ++nt) {
      v8f c = {0.f, 0.f, 0.f, 0.f, 0.f, 0.f, 0.f, 0.f};
      for (int kt = 0; kt < KTN; ++kt) {
        v16h a = load_a(inRow + kt * 32, KPAD, lane);
        v16h b = load_b_packed(Wp + (size_t)(kt * NTN + nt) * 512, lane);
        c = wmma_acc(a, b, c);
      }
      const int n = nt * 16 + col;
      const float bv = (n < realN) ? bias[n] : 0.0f;
#pragma unroll
      for (int v = 0; v < 8; ++v) {
        float x = c[v] + bv;
        x = fmaxf(x, 0.0f);
        outLds[(mt * 16 + rb + v) * NPAD + n] = (_Float16)x;
      }
    }
  }
}

// Tail GEMM: [rows][64] f16 @ W3 (64x16 padded) -> f32 accumulator fragment.
__device__ inline v8f mlp_tail64(const _Float16* __restrict__ inLds, int rowOff,
                                 const _Float16* __restrict__ W3p, int lane) {
  v8f c = {0.f, 0.f, 0.f, 0.f, 0.f, 0.f, 0.f, 0.f};
#pragma unroll
  for (int kt = 0; kt < 2; ++kt) {
    v16h a = load_a(inLds + rowOff * 64 + kt * 32, 64, lane);
    v16h b = load_b_packed(W3p + (size_t)kt * 512, lane);
    c = wmma_acc(a, b, c);
  }
  return c;
}

// ---------------------------------------------------------------------------
// Weight conversion/packing: fp32 [K][N] row-major -> f16 fragment-major
// [Kpad/32][Npad/16][512], zero padded.
// ---------------------------------------------------------------------------
__global__ void cvt_pack_kernel(const float* __restrict__ W,
                                _Float16* __restrict__ dst, int K, int N,
                                int Kpad, int Npad) {
  const int total = Kpad * Npad;
  const int ntn = Npad >> 4;
  for (int idx = blockIdx.x * blockDim.x + threadIdx.x; idx < total;
       idx += gridDim.x * blockDim.x) {
    const int tile = idx >> 9;
    const int r = idx & 511;
    const int lane = r >> 4;
    const int j = r & 15;
    const int kt = tile / ntn;
    const int nt = tile - kt * ntn;
    const int col = lane & 15;
    const int kb = (lane >> 4) << 3;
    const int kk = (j < 8) ? (kb + j) : (16 + kb + (j - 8));
    const int k = kt * 32 + kk;
    const int n = nt * 16 + col;
    dst[idx] = (k < K && n < N) ? (_Float16)W[(size_t)k * N + n] : (_Float16)0.f;
  }
}

// ---------------------------------------------------------------------------
// h[:, 0:8] = embed[Z]   (h stored [natom][32] = concat of h0..h3)
// ---------------------------------------------------------------------------
__global__ void embed_kernel(const int* __restrict__ Z,
                             const float* __restrict__ emb,
                             float* __restrict__ h, int natom) {
  int i = blockIdx.x * blockDim.x + threadIdx.x;
  if (i >= natom * 8) return;
  int a = i >> 3, k = i & 7;
  h[(size_t)a * 32 + k] = emb[(size_t)Z[a] * 8 + k];
}

// ---------------------------------------------------------------------------
// Distances + unit vectors + envelope*sin RBF (+ optional 1/d^3)
// ---------------------------------------------------------------------------
__global__ void geom_kernel(const float* __restrict__ Ra,
                            const float* __restrict__ Rb,
                            const int* __restrict__ src,
                            const int* __restrict__ tgt,
                            const float* __restrict__ freqs, float rcutInv,
                            float* __restrict__ rbf, float* __restrict__ unit,
                            float* __restrict__ inv3, int ne) {
  int e = blockIdx.x * blockDim.x + threadIdx.x;
  if (e >= ne) return;
  int s = src[e], t = tgt[e];
  float dx = Rb[(size_t)t * 3 + 0] - Ra[(size_t)s * 3 + 0];
  float dy = Rb[(size_t)t * 3 + 1] - Ra[(size_t)s * 3 + 1];
  float dz = Rb[(size_t)t * 3 + 2] - Ra[(size_t)s * 3 + 2];
  float d2 = fmaxf(dx * dx + dy * dy + dz * dz, 1e-10f);
  float d = sqrtf(d2);
  float ood = 1.0f / d;
  unit[(size_t)e * 3 + 0] = dx * ood;
  unit[(size_t)e * 3 + 1] = dy * ood;
  unit[(size_t)e * 3 + 2] = dz * ood;
  float x = d * rcutInv;
  float env = 0.0f;
  if (x < 1.0f) {
    float x2 = x * x;
    float x5 = x2 * x2 * x;
    float x6 = x5 * x;
    float x7 = x6 * x;
    env = 1.0f / x + (-28.0f) * x5 + 48.0f * x6 + (-21.0f) * x7;
  }
#pragma unroll
  for (int k = 0; k < 8; ++k)
    rbf[(size_t)e * 8 + k] = env * sinf(freqs[k] * x);
  if (inv3) inv3[e] = ood * ood * ood;
}

// ---------------------------------------------------------------------------
// Message features + segment-sum at source atom via float atomics.
// m_ij layout: [h_all(32) | h_all (x) rbf (256) | rbf(8)] = 296
// ---------------------------------------------------------------------------
__global__ void msg_agg_kernel(const float* __restrict__ h, int hprevOff,
                               const float* __restrict__ rbf,
                               const int* __restrict__ src,
                               const int* __restrict__ tgt,
                               float* __restrict__ m, int ne) {
  int e = blockIdx.x * blockDim.x + threadIdx.x;
  if (e >= ne) return;
  int s = src[e], t = tgt[e];
  float hall[32], rb[8];
#pragma unroll
  for (int k = 0; k < 8; ++k) {
    hall[k] = h[(size_t)s * 32 + k];
    hall[8 + k] = h[(size_t)t * 32 + k];
    hall[16 + k] = h[(size_t)s * 32 + hprevOff + k];
    hall[24 + k] = h[(size_t)t * 32 + hprevOff + k];
    rb[k] = rbf[(size_t)e * 8 + k];
  }
  float* ms = m + (size_t)s * 296;
#pragma unroll
  for (int a = 0; a < 32; ++a) atomicAdd(&ms[a], hall[a]);
  for (int a = 0; a < 32; ++a) {
#pragma unroll
    for (int r = 0; r < 8; ++r) atomicAdd(&ms[32 + a * 8 + r], hall[a] * rb[r]);
  }
#pragma unroll
  for (int r = 0; r < 8; ++r) atomicAdd(&ms[288 + r], rb[r]);
}

// ---------------------------------------------------------------------------
// Update MLP: m[natom][296] -> h[:, outOff:outOff+8].  One wave per 16 atoms.
// ---------------------------------------------------------------------------
__global__ __launch_bounds__(32) void update_mlp_kernel(
    const float* __restrict__ m, const _Float16* __restrict__ Wp,
    const float* __restrict__ b0, const float* __restrict__ b1,
    const float* __restrict__ b2, const float* __restrict__ b3,
    float* __restrict__ hOut, int outOff, int natom) {
  __shared__ __align__(16) _Float16 a0[16 * 320];
  __shared__ __align__(16) _Float16 a1[16 * 256];
  __shared__ __align__(16) _Float16 a2[16 * 128];
  __shared__ __align__(16) _Float16 a3[16 * 64];
  const int lane = threadIdx.x;
  const int base = blockIdx.x * 16;
  for (int idx = lane; idx < 16 * 320; idx += 32) {
    int r = idx / 320, f = idx - r * 320;
    int at = base + r;
    float v = (f < 296 && at < natom) ? m[(size_t)at * 296 + f] : 0.0f;
    a0[idx] = (_Float16)v;
  }
  __syncthreads();
  mlp_layer<320, 256, 1>(a0, Wp, b0, 256, a1, lane);
  __syncthreads();
  mlp_layer<256, 128, 1>(a1, Wp + HID_W0, b1, 128, a2, lane);
  __syncthreads();
  mlp_layer<128, 64, 1>(a2, Wp + HID_W0 + HID_W1, b2, 64, a3, lane);
  __syncthreads();
  v8f c = mlp_tail64(a3, 0, Wp + HID_W0 + HID_W1 + HID_W2, lane);
  const int col = lane & 15;
  const int rb = (lane >> 4) << 3;
  if (col < 8) {
    float bv = b3[col];
#pragma unroll
    for (int v = 0; v < 8; ++v) {
      int at = base + rb + v;
      if (at < natom) hOut[(size_t)at * 32 + outOff + col] = c[v] + bv;
    }
  }
}

// ---------------------------------------------------------------------------
// Directional MLP on edges: rebuild m_ij on the fly, MLP, scale by unit vector
// and segment-sum (atomics) into hdir[natom][3][24] at column dirOff.
// ---------------------------------------------------------------------------
__global__ __launch_bounds__(32) void dir_mlp_kernel(
    const float* __restrict__ h, int hprevOff, const float* __restrict__ rbf,
    const float* __restrict__ unit, const int* __restrict__ src,
    const int* __restrict__ tgt, const _Float16* __restrict__ Wp,
    const float* __restrict__ b0, const float* __restrict__ b1,
    const float* __restrict__ b2, const float* __restrict__ b3,
    float* __restrict__ hdir, int dirOff, int ne) {
  __shared__ float hallS[16 * 32];
  __shared__ float rbS[16 * 8];
  __shared__ __align__(16) _Float16 a0[16 * 320];
  __shared__ __align__(16) _Float16 a1[16 * 256];
  __shared__ __align__(16) _Float16 a2[16 * 128];
  __shared__ __align__(16) _Float16 a3[16 * 64];
  const int lane = threadIdx.x;
  const int base = blockIdx.x * 16;
  for (int idx = lane; idx < 16 * 32; idx += 32) {
    int r = idx >> 5, k = idx & 31;
    int e = base + r;
    float v = 0.0f;
    if (e < ne) {
      int kk = k & 7;
      int sel = k >> 3;
      int at = (sel == 0 || sel == 2) ? src[e] : tgt[e];
      int off = (sel < 2) ? kk : (hprevOff + kk);
      v = h[(size_t)at * 32 + off];
    }
    hallS[idx] = v;
  }
  for (int idx = lane; idx < 16 * 8; idx += 32) {
    int r = idx >> 3, k = idx & 7;
    int e = base + r;
    rbS[idx] = (e < ne) ? rbf[(size_t)e * 8 + k] : 0.0f;
  }
  __syncthreads();
  for (int idx = lane; idx < 16 * 320; idx += 32) {
    int r = idx / 320, f = idx - r * 320;
    float v;
    if (f < 32) {
      v = hallS[r * 32 + f];
    } else if (f < 288) {
      int a = (f - 32) >> 3, q = (f - 32) & 7;
      v = hallS[r * 32 + a] * rbS[r * 8 + q];
    } else if (f < 296) {
      v = rbS[r * 8 + (f - 288)];
    } else {
      v = 0.0f;
    }
    a0[idx] = (_Float16)v;
  }
  __syncthreads();
  mlp_layer<320, 256, 1>(a0, Wp, b0, 256, a1, lane);
  __syncthreads();
  mlp_layer<256, 128, 1>(a1, Wp + HID_W0, b1, 128, a2, lane);
  __syncthreads();
  mlp_layer<128, 64, 1>(a2, Wp + HID_W0 + HID_W1, b2, 64, a3, lane);
  __syncthreads();
  v8f c = mlp_tail64(a3, 0, Wp + HID_W0 + HID_W1 + HID_W2, lane);
  const int col = lane & 15;
  const int rb = (lane >> 4) << 3;
  if (col < 8) {
    float bv = b3[col];
#pragma unroll
    for (int v = 0; v < 8; ++v) {
      int e = base + rb + v;
      if (e < ne) {
        float val = c[v] + bv;
        int s = src[e];
        float ux = unit[(size_t)e * 3 + 0];
        float uy = unit[(size_t)e * 3 + 1];
        float uz = unit[(size_t)e * 3 + 2];
        float* dst = hdir + (size_t)s * 72 + dirOff + col;
        atomicAdd(dst + 0, ux * val);
        atomicAdd(dst + 24, uy * val);
        atomicAdd(dst + 48, uz * val);
      }
    }
  }
}

// ---------------------------------------------------------------------------
// Pair features (hAB rows 0-15, hBA rows 16-31) + 4 readout heads,
// E = (ro(hAB)+ro(hBA)) / d^3, atomically summed per dimer.
// ---------------------------------------------------------------------------
__global__ __launch_bounds__(32) void pair_readout_kernel(
    const float* __restrict__ hA, const float* __restrict__ hB,
    const float* __restrict__ qA, const float* __restrict__ qB,
    const float* __restrict__ hAdir, const float* __restrict__ hBdir,
    const float* __restrict__ rbf, const float* __restrict__ unit,
    const float* __restrict__ inv3, const int* __restrict__ src,
    const int* __restrict__ tgt, const int* __restrict__ dimer,
    const _Float16* __restrict__ wRo, RoArgs ra, float* __restrict__ out,
    int ne) {
  __shared__ __align__(16) _Float16 a0[32 * 128];
  __shared__ __align__(16) _Float16 a1[32 * 256];
  __shared__ __align__(16) _Float16 a2[32 * 128];
  __shared__ __align__(16) _Float16 a3[32 * 64];
  __shared__ float proj[16 * 48];
  __shared__ float outv[32];
  const int lane = threadIdx.x;
  const int base = blockIdx.x * 16;
  // Directional projections: Aproj = hAdir . u ; Bproj = hBdir . (-u)
  for (int idx = lane; idx < 16 * 48; idx += 32) {
    int r = idx / 48, j = idx - r * 48;
    int e = base + r;
    float v = 0.0f;
    if (e < ne) {
      int f = j % 24;
      bool isB = j >= 24;
      int at = isB ? tgt[e] : src[e];
      const float* hd = isB ? hBdir : hAdir;
      float sgn = isB ? -1.0f : 1.0f;
      float ux = unit[(size_t)e * 3 + 0];
      float uy = unit[(size_t)e * 3 + 1];
      float uz = unit[(size_t)e * 3 + 2];
      v = sgn * (hd[(size_t)at * 72 + f] * ux + hd[(size_t)at * 72 + 24 + f] * uy +
                 hd[(size_t)at * 72 + 48 + f] * uz);
    }
    proj[idx] = v;
  }
  __syncthreads();
  // Build 122-wide features (padded to 128) for hAB and hBA.
  for (int idx = lane; idx < 32 * 128; idx += 32) {
    int row = idx >> 7, f = idx & 127;
    int r = row & 15;
    bool ba = row >= 16;
    int e = base + r;
    float v = 0.0f;
    if (e < ne) {
      int s = src[e], t = tgt[e];
      if (f < 32)
        v = ba ? hB[(size_t)t * 32 + f] : hA[(size_t)s * 32 + f];
      else if (f < 64)
        v = ba ? hA[(size_t)s * 32 + (f - 32)] : hB[(size_t)t * 32 + (f - 32)];
      else if (f == 64)
        v = ba ? qB[t] : qA[s];
      else if (f == 65)
        v = ba ? qA[s] : qB[t];
      else if (f < 74)
        v = rbf[(size_t)e * 8 + (f - 66)];
      else if (f < 98)
        v = proj[r * 48 + (ba ? 24 : 0) + (f - 74)];
      else if (f < 122)
        v = proj[r * 48 + (ba ? 0 : 24) + (f - 98)];
    }
    a0[idx] = (_Float16)v;
  }
  __syncthreads();
  const int col = lane & 15;
  const int rbl = (lane >> 4) << 3;
  for (int head = 0; head < 4; ++head) {
    const _Float16* W = wRo + (size_t)head * RO_TOT;
    mlp_layer<128, 256, 2>(a0, W, ra.b0[head], 256, a1, lane);
    __syncthreads();
    mlp_layer<256, 128, 2>(a1, W + RO_W0, ra.b1[head], 128, a2, lane);
    __syncthreads();
    mlp_layer<128, 64, 2>(a2, W + RO_W0 + RO_W1, ra.b2[head], 64, a3, lane);
    __syncthreads();
    const _Float16* W3 = W + RO_W0 + RO_W1 + RO_W2;
    const float b3v = ra.b3[head][0];
    for (int mt = 0; mt < 2; ++mt) {
      v8f c = mlp_tail64(a3, mt * 16, W3, lane);
      if (col == 0) {
#pragma unroll
        for (int v = 0; v < 8; ++v) outv[mt * 16 + rbl + v] = c[v] + b3v;
      }
    }
    __syncthreads();
    if (lane < 16) {
      int e = base + lane;
      if (e < ne) {
        float E = (outv[lane] + outv[lane + 16]) * inv3[e];
        atomicAdd(&out[(size_t)dimer[e] * 4 + head], E);
      }
    }
    __syncthreads();
  }
}

// ---------------------------------------------------------------------------
// Long-range multipole electrostatics -> out[:,0]
// ---------------------------------------------------------------------------
__global__ void mtp_kernel(const float* __restrict__ RA,
                           const float* __restrict__ RB,
                           const float* __restrict__ qA,
                           const float* __restrict__ muA,
                           const float* __restrict__ quadA,
                           const float* __restrict__ qB,
                           const float* __restrict__ muB,
                           const float* __restrict__ quadB,
                           const int* __restrict__ src,
                           const int* __restrict__ tgt,
                           const int* __restrict__ dimer,
                           float* __restrict__ out, int ne) {
  int e = blockIdx.x * blockDim.x + threadIdx.x;
  if (e >= ne) return;
  const float B2A = 1.0f / 0.529177210903f;
  int s = src[e], t = tgt[e];
  float dx = RB[(size_t)t * 3 + 0] - RA[(size_t)s * 3 + 0];
  float dy = RB[(size_t)t * 3 + 1] - RA[(size_t)s * 3 + 1];
  float dz = RB[(size_t)t * 3 + 2] - RA[(size_t)s * 3 + 2];
  float d2 = fmaxf(dx * dx + dy * dy + dz * dz, 1e-10f);
  float dAng = sqrtf(d2);
  float dR = dAng * B2A;
  float r[3] = {dx * B2A, dy * B2A, dz * B2A};
  float oo = 1.0f / dR;
  float oo3 = oo * oo * oo;
  float oo5 = oo3 * oo * oo;
  float qa = qA[s], qb = qB[t];
  float mA[3], mB[3];
#pragma unroll
  for (int k = 0; k < 3; ++k) {
    mA[k] = muA[(size_t)s * 3 + k];
    mB[k] = muB[(size_t)t * 3 + k];
  }
  float E = qa * qb * oo;
  // charge-dipole
#pragma unroll
  for (int k = 0; k < 3; ++k) {
    float T1 = -oo3 * r[k];
    E += T1 * (qa * mB[k] - qb * mA[k]);
  }
  // T2 contractions
  float dRdR = dR * dR;
  float euu = 0.0f, eqQ = 0.0f;
#pragma unroll
  for (int y = 0; y < 3; ++y) {
#pragma unroll
    for (int z = 0; z < 3; ++z) {
      float T2 = oo5 * (3.0f * r[y] * r[z] - ((y == z) ? dRdR : 0.0f));
      euu += mA[y] * mB[z] * T2;
      eqQ += T2 * (qa * quadB[(size_t)t * 9 + y * 3 + z] +
                   qb * quadA[(size_t)s * 9 + y * 3 + z]);
    }
  }
  E += -euu + eqQ * (1.0f / 3.0f);
  atomicAdd(&out[(size_t)dimer[e] * 4 + 0], 627.509f * E);
}

// ---------------------------------------------------------------------------
// Host orchestration
// ---------------------------------------------------------------------------
extern "C" void kernel_launch(void* const* d_in, const int* in_sizes, int n_in,
                              void* d_out, int out_size, void* d_ws,
                              size_t ws_size, hipStream_t stream) {
  (void)n_in;
  (void)out_size;
  (void)ws_size;
  const int natomA = in_sizes[0];
  const int natomB = in_sizes[2];
  const int nsr = in_sizes[4];
  const int nlr = in_sizes[7];
  const int nAA = in_sizes[10];
  const int nBB = in_sizes[12];
  const int ndimer = in_sizes[14];

  const int* ZA = (const int*)d_in[0];
  const float* RA = (const float*)d_in[1];
  const int* ZB = (const int*)d_in[2];
  const float* RB = (const float*)d_in[3];
  const int* sr_s = (const int*)d_in[4];
  const int* sr_t = (const int*)d_in[5];
  const int* sr_d = (const int*)d_in[6];
  const int* lr_s = (const int*)d_in[7];
  const int* lr_t = (const int*)d_in[8];
  const int* lr_d = (const int*)d_in[9];
  const int* aa_s = (const int*)d_in[10];
  const int* aa_t = (const int*)d_in[11];
  const int* bb_s = (const int*)d_in[12];
  const int* bb_t = (const int*)d_in[13];
  const float* qA = (const float*)d_in[16];
  const float* muA = (const float*)d_in[17];
  const float* quadA = (const float*)d_in[18];
  const float* qB = (const float*)d_in[20];
  const float* muB = (const float*)d_in[21];
  const float* quadB = (const float*)d_in[22];
  // params leaves (tree-flatten, sorted keys):
  // 24..47 directional[0..2] (W0,b0,W1,b1,W2,b2,W3,b3), 48 embed, 49 freq,
  // 50 freq_im, 51..58 ro_disp, 59..66 ro_elst, 67..74 ro_exch,
  // 75..82 ro_indu, 83..106 update[0..2]
  const float* embedW = (const float*)d_in[48];
  const float* freqs = (const float*)d_in[49];
  const float* freqs_im = (const float*)d_in[50];
  float* out = (float*)d_out;

  // ---- workspace layout -------------------------------------------------
  char* ws = (char*)d_ws;
  size_t cur = 0;
  auto alloc = [&](size_t bytes) -> void* {
    cur = (cur + 255) & ~(size_t)255;
    void* p = ws + cur;
    cur += bytes;
    return p;
  };
  _Float16* wUpd[3];
  _Float16* wDir[3];
  for (int i = 0; i < 3; ++i) wUpd[i] = (_Float16*)alloc(sizeof(_Float16) * HID_TOT);
  for (int i = 0; i < 3; ++i) wDir[i] = (_Float16*)alloc(sizeof(_Float16) * HID_TOT);
  _Float16* wRo = (_Float16*)alloc(sizeof(_Float16) * (size_t)RO_TOT * 4);
  float* hA = (float*)alloc(sizeof(float) * (size_t)natomA * 32);
  float* hB = (float*)alloc(sizeof(float) * (size_t)natomB * 32);
  float* hAdir = (float*)alloc(sizeof(float) * (size_t)natomA * 72);
  float* hBdir = (float*)alloc(sizeof(float) * (size_t)natomB * 72);
  float* mA = (float*)alloc(sizeof(float) * (size_t)natomA * 296);
  float* mB = (float*)alloc(sizeof(float) * (size_t)natomB * 296);
  float* rbfAA = (float*)alloc(sizeof(float) * (size_t)nAA * 8);
  float* unitAA = (float*)alloc(sizeof(float) * (size_t)nAA * 3);
  float* rbfBB = (float*)alloc(sizeof(float) * (size_t)nBB * 8);
  float* unitBB = (float*)alloc(sizeof(float) * (size_t)nBB * 3);
  float* rbfSR = (float*)alloc(sizeof(float) * (size_t)nsr * 8);
  float* unitSR = (float*)alloc(sizeof(float) * (size_t)nsr * 3);
  float* inv3SR = (float*)alloc(sizeof(float) * (size_t)nsr);

  // ---- weight conversion / packing -------------------------------------
  auto cvt = [&](int idx, _Float16* dst, int K, int N, int Kpad, int Npad) {
    int total = Kpad * Npad;
    cvt_pack_kernel<<<(total + 255) / 256, 256, 0, stream>>>(
        (const float*)d_in[idx], dst, K, N, Kpad, Npad);
  };
  for (int i = 0; i < 3; ++i) {
    int ub = 83 + i * 8;
    int db = 24 + i * 8;
    cvt(ub + 0, wUpd[i], 296, 256, 320, 256);
    cvt(ub + 2, wUpd[i] + HID_W0, 256, 128, 256, 128);
    cvt(ub + 4, wUpd[i] + HID_W0 + HID_W1, 128, 64, 128, 64);
    cvt(ub + 6, wUpd[i] + HID_W0 + HID_W1 + HID_W2, 64, 8, 64, 16);
    cvt(db + 0, wDir[i], 296, 256, 320, 256);
    cvt(db + 2, wDir[i] + HID_W0, 256, 128, 256, 128);
    cvt(db + 4, wDir[i] + HID_W0 + HID_W1, 128, 64, 128, 64);
    cvt(db + 6, wDir[i] + HID_W0 + HID_W1 + HID_W2, 64, 8, 64, 16);
  }
  const int roBase[4] = {59, 67, 75, 51};  // elst, exch, indu, disp -> cols 0..3
  RoArgs ra;
  for (int h = 0; h < 4; ++h) {
    _Float16* w = wRo + (size_t)h * RO_TOT;
    cvt(roBase[h] + 0, w, 122, 256, 128, 256);
    cvt(roBase[h] + 2, w + RO_W0, 256, 128, 256, 128);
    cvt(roBase[h] + 4, w + RO_W0 + RO_W1, 128, 64, 128, 64);
    cvt(roBase[h] + 6, w + RO_W0 + RO_W1 + RO_W2, 64, 1, 64, 16);
    ra.b0[h] = (const float*)d_in[roBase[h] + 1];
    ra.b1[h] = (const float*)d_in[roBase[h] + 3];
    ra.b2[h] = (const float*)d_in[roBase[h] + 5];
    ra.b3[h] = (const float*)d_in[roBase[h] + 7];
  }

  // ---- init accumulators -------------------------------------------------
  hipMemsetAsync(out, 0, sizeof(float) * (size_t)ndimer * 4, stream);
  hipMemsetAsync(hAdir, 0, sizeof(float) * (size_t)natomA * 72, stream);
  hipMemsetAsync(hBdir, 0, sizeof(float) * (size_t)natomB * 72, stream);

  // ---- embeddings + geometry --------------------------------------------
  embed_kernel<<<(natomA * 8 + 255) / 256, 256, 0, stream>>>(ZA, embedW, hA, natomA);
  embed_kernel<<<(natomB * 8 + 255) / 256, 256, 0, stream>>>(ZB, embedW, hB, natomB);
  geom_kernel<<<(nAA + 255) / 256, 256, 0, stream>>>(RA, RA, aa_s, aa_t, freqs,
                                                     0.2f, rbfAA, unitAA, nullptr, nAA);
  geom_kernel<<<(nBB + 255) / 256, 256, 0, stream>>>(RB, RB, bb_s, bb_t, freqs,
                                                     0.2f, rbfBB, unitBB, nullptr, nBB);
  geom_kernel<<<(nsr + 255) / 256, 256, 0, stream>>>(RA, RB, sr_s, sr_t, freqs_im,
                                                     0.125f, rbfSR, unitSR, inv3SR, nsr);

  // ---- message passing iterations ---------------------------------------
  for (int it = 0; it < 3; ++it) {
    const int hprev = 8 * it;
    const int hout = 8 * (it + 1);
    const int doff = 8 * it;
    const int ub = 83 + it * 8;
    const int db = 24 + it * 8;
    hipMemsetAsync(mA, 0, sizeof(float) * (size_t)natomA * 296, stream);
    hipMemsetAsync(mB, 0, sizeof(float) * (size_t)natomB * 296, stream);
    msg_agg_kernel<<<(nAA + 255) / 256, 256, 0, stream>>>(hA, hprev, rbfAA, aa_s,
                                                          aa_t, mA, nAA);
    msg_agg_kernel<<<(nBB + 255) / 256, 256, 0, stream>>>(hB, hprev, rbfBB, bb_s,
                                                          bb_t, mB, nBB);
    update_mlp_kernel<<<(natomA + 15) / 16, 32, 0, stream>>>(
        mA, wUpd[it], (const float*)d_in[ub + 1], (const float*)d_in[ub + 3],
        (const float*)d_in[ub + 5], (const float*)d_in[ub + 7], hA, hout, natomA);
    update_mlp_kernel<<<(natomB + 15) / 16, 32, 0, stream>>>(
        mB, wUpd[it], (const float*)d_in[ub + 1], (const float*)d_in[ub + 3],
        (const float*)d_in[ub + 5], (const float*)d_in[ub + 7], hB, hout, natomB);
    dir_mlp_kernel<<<(nAA + 15) / 16, 32, 0, stream>>>(
        hA, hprev, rbfAA, unitAA, aa_s, aa_t, wDir[it],
        (const float*)d_in[db + 1], (const float*)d_in[db + 3],
        (const float*)d_in[db + 5], (const float*)d_in[db + 7], hAdir, doff, nAA);
    dir_mlp_kernel<<<(nBB + 15) / 16, 32, 0, stream>>>(
        hB, hprev, rbfBB, unitBB, bb_s, bb_t, wDir[it],
        (const float*)d_in[db + 1], (const float*)d_in[db + 3],
        (const float*)d_in[db + 5], (const float*)d_in[db + 7], hBdir, doff, nBB);
  }

  // ---- short-range readout + long-range electrostatics -------------------
  pair_readout_kernel<<<(nsr + 15) / 16, 32, 0, stream>>>(
      hA, hB, qA, qB, hAdir, hBdir, rbfSR, unitSR, inv3SR, sr_s, sr_t, sr_d, wRo,
      ra, out, nsr);
  mtp_kernel<<<(nlr + 255) / 256, 256, 0, stream>>>(RA, RB, qA, muA, quadA, qB,
                                                    muB, quadB, lr_s, lr_t, lr_d,
                                                    out, nlr);
}
